// HierarchicalNeighborNetwork_2104533975109
// MI455X (gfx1250) — compile-verified, matrix operand order checked by reading, not run on hardware
//
#include <hip/hip_runtime.h>
#include <hip/hip_bf16.h>

#define N_NODES 100000
#define N_EDGES 1000000
#define IN_C  64
#define HID_C 128
#define OUT_C 40
#define OUT_PAD 48   // 40 padded to a multiple of 16

typedef __bf16 v16bf __attribute__((ext_vector_type(16)));
typedef __bf16 v8bf  __attribute__((ext_vector_type(8)));
typedef float  v8f   __attribute__((ext_vector_type(8)));

// f32 -> bf16 bits with round-to-nearest-even
__device__ __forceinline__ unsigned int bfbits(float f) {
    unsigned int u = __builtin_bit_cast(unsigned int, f);
    u += 0x7FFFu + ((u >> 16) & 1u);
    return u >> 16;
}
__device__ __forceinline__ __bf16 f2bf(float f) {
    unsigned short h = (unsigned short)bfbits(f);
    return __builtin_bit_cast(__bf16, h);
}

__global__ void k_zero(float* __restrict__ p, long long n) {
    long long i = (long long)blockIdx.x * blockDim.x + threadIdx.x;
    long long stride = (long long)gridDim.x * blockDim.x;
    for (; i < n; i += stride) p[i] = 0.0f;
}

__global__ void k_deg(const long long* __restrict__ dst, float* __restrict__ deg, int e) {
    int i = blockIdx.x * blockDim.x + threadIdx.x;
    if (i < e) atomicAdd(&deg[(int)dst[i]], 1.0f);
}

__global__ void k_rdeg(float* __restrict__ deg, int n) {
    int i = blockIdx.x * blockDim.x + threadIdx.x;
    if (i < n) deg[i] = 1.0f / fmaxf(deg[i], 1.0f);
}

// scatter-add edge messages: agg[dst] += feat[src]; thread <-> (edge, channel):
// consecutive lanes hit consecutive channels; 51MB agg target is L2-resident.
__global__ void k_scatter(const float* __restrict__ feat,
                          const long long* __restrict__ src,
                          const long long* __restrict__ dst,
                          float* __restrict__ agg, int C) {
    long long total = (long long)N_EDGES * C;
    long long i = (long long)blockIdx.x * blockDim.x + threadIdx.x;
    long long stride = (long long)gridDim.x * blockDim.x;
    for (; i < total; i += stride) {
        int e = (int)(i / C);
        int c = (int)(i - (long long)e * C);
        int s = (int)src[e];
        int d = (int)dst[e];
        atomicAdd(&agg[(long long)d * C + c], feat[(long long)s * C + c]);
    }
}

// pack (agg*rdeg) and root features to bf16 operand buffers, 2 elems/thread (b32 stores)
__global__ void k_cvt_pair(const float* __restrict__ agg, const float* __restrict__ rdeg,
                           const float* __restrict__ xsrc,
                           unsigned int* __restrict__ aggBf, unsigned int* __restrict__ xBf,
                           int K, long long nPairs) {
    long long i = (long long)blockIdx.x * blockDim.x + threadIdx.x;
    long long stride = (long long)gridDim.x * blockDim.x;
    for (; i < nPairs; i += stride) {
        long long e0 = i * 2;
        int row = (int)(e0 / K);
        float rd = rdeg[row];
        float a0 = agg[e0] * rd, a1 = agg[e0 + 1] * rd;
        float x0 = xsrc[e0],     x1 = xsrc[e0 + 1];
        aggBf[i] = bfbits(a0) | (bfbits(a1) << 16);
        xBf[i]   = bfbits(x0) | (bfbits(x1) << 16);
    }
}

// one-time: W[K][Nout] f32 -> W^T[Npad][K] bf16, zero-padded rows
__global__ void k_wt(const float* __restrict__ w, __bf16* __restrict__ wt,
                     int K, int Nout, int Npad) {
    int i = blockIdx.x * blockDim.x + threadIdx.x;
    int total = Npad * K;
    if (i >= total) return;
    int n = i / K, k = i - n * K;
    float v = (n < Nout) ? w[(long long)k * Nout + n] : 0.0f;
    wt[i] = f2bf(v);
}

__global__ void k_biaspad(const float* __restrict__ b, float* __restrict__ bp,
                          int Nout, int Npad) {
    int i = blockIdx.x * blockDim.x + threadIdx.x;
    if (i < Npad) bp[i] = (i < Nout) ? b[i] : 0.0f;
}

// Fragment load: ISA 7.12.2 16-bit A/B layout. Lane's 16 bf16 are two contiguous
// 16B runs: k in [half*8, half*8+8) and [16+half*8, 16+half*8+8)  -> 2x b128 loads.
__device__ __forceinline__ v16bf frag(const __bf16* __restrict__ p, int half) {
    v8bf lo = *(const v8bf*)(p + half * 8);
    v8bf hi = *(const v8bf*)(p + 16 + half * 8);
    return __builtin_shufflevector(lo, hi, 0, 1, 2, 3, 4, 5, 6, 7,
                                           8, 9, 10, 11, 12, 13, 14, 15);
}

// out = (agg*rdeg)@wl + x@wr + bias [+ReLU]; one wave per 16x16 tile, K unrolled.
template <int K>
__global__ void k_gemm(const __bf16* __restrict__ aggBf, const __bf16* __restrict__ xBf,
                       const __bf16* __restrict__ wtl, const __bf16* __restrict__ wtr,
                       const float* __restrict__ biasPad, float* __restrict__ out,
                       int Nout, int nTilesN, int totalTiles, int relu) {
    int wave = (blockIdx.x * blockDim.x + threadIdx.x) >> 5;
    if (wave >= totalTiles) return;          // wave-uniform, EXEC all-1 for WMMA
    int lane = threadIdx.x & 31;
    int half = lane >> 4;
    int idx  = lane & 15;
    int mt = wave / nTilesN;
    int nt = wave - mt * nTilesN;
    int row  = mt * 16 + idx;                // A row owned by this lane
    int ncol = nt * 16 + idx;                // B/D column (always < Npad)
    const __bf16* aRow = aggBf + (long long)row * K;
    const __bf16* xRow = xBf   + (long long)row * K;
    const __bf16* lRow = wtl   + (long long)ncol * K;   // W^T row = column of W
    const __bf16* rRow = wtr   + (long long)ncol * K;

    v8f c = {};
#pragma unroll
    for (int kk = 0; kk < K; kk += 32) {
        v16bf a1 = frag(aRow + kk, half);
        v16bf b1 = frag(lRow + kk, half);
        v16bf a2 = frag(xRow + kk, half);
        v16bf b2 = frag(rRow + kk, half);
        c = __builtin_amdgcn_wmma_f32_16x16x32_bf16(false, a1, false, b1,
                                                    (short)0, c, false, false);
        c = __builtin_amdgcn_wmma_f32_16x16x32_bf16(false, a2, false, b2,
                                                    (short)0, c, false, false);
    }

    float bv = biasPad[ncol];
    if (ncol < Nout) {                        // only the 40-col layer is ragged
#pragma unroll
        for (int r = 0; r < 8; ++r) {
            int orow = mt * 16 + half * 8 + r;   // C/D layout: row = vgpr + half*8
            float v = c[r] + bv;
            if (relu) v = fmaxf(v, 0.0f);
            out[(long long)orow * Nout + ncol] = v;
        }
    }
}

// in-place row-wise log_softmax, one wave32 per row
__global__ void k_logsoftmax(float* __restrict__ out, int n, int C) {
    int wave = (blockIdx.x * blockDim.x + threadIdx.x) >> 5;
    int lane = threadIdx.x & 31;
    int nw = (gridDim.x * blockDim.x) >> 5;
    for (int r = wave; r < n; r += nw) {
        float* rowp = out + (long long)r * C;
        float v0 = (lane      < C) ? rowp[lane]      : -INFINITY;
        float v1 = (lane + 32 < C) ? rowp[lane + 32] : -INFINITY;
        float m = fmaxf(v0, v1);
        for (int off = 16; off > 0; off >>= 1)
            m = fmaxf(m, __shfl_xor(m, off, 32));
        float s = 0.0f;
        if (lane      < C) s += __expf(v0 - m);
        if (lane + 32 < C) s += __expf(v1 - m);
        for (int off = 16; off > 0; off >>= 1)
            s += __shfl_xor(s, off, 32);
        float ls = __logf(s);
        if (lane      < C) rowp[lane]      = v0 - m - ls;
        if (lane + 32 < C) rowp[lane + 32] = v1 - m - ls;
    }
}

extern "C" void kernel_launch(void* const* d_in, const int* in_sizes, int n_in,
                              void* d_out, int out_size, void* d_ws, size_t ws_size,
                              hipStream_t stream) {
    (void)in_sizes; (void)n_in; (void)out_size; (void)ws_size;
    const float*     x   = (const float*)d_in[0];
    const long long* ei  = (const long long*)d_in[1];
    const float*     w1l = (const float*)d_in[2];
    const float*     w1r = (const float*)d_in[3];
    const float*     b1  = (const float*)d_in[4];
    const float*     w2l = (const float*)d_in[5];
    const float*     w2r = (const float*)d_in[6];
    const float*     b2  = (const float*)d_in[7];
    const float*     w3l = (const float*)d_in[8];
    const float*     w3r = (const float*)d_in[9];
    const float*     b3  = (const float*)d_in[10];
    float* out = (float*)d_out;

    const long long* src = ei;
    const long long* dst = ei + N_EDGES;

    // ---- workspace layout ----
    // f32 region: rdeg[N] | agg[N*128] | h1[N*128] | h2[N*128] | biasPad[3*128]
    // bf16 region: aggBf[N*128] | xBf[N*128] | wt1l,wt1r[128*64] | wt2l,wt2r[128*128] | wt3l,wt3r[48*128]
    float* deg = (float*)d_ws;
    float* agg = deg + N_NODES;
    float* h1  = agg + (long long)N_NODES * HID_C;
    float* h2  = h1  + (long long)N_NODES * HID_C;
    float* bp1 = h2  + (long long)N_NODES * HID_C;
    float* bp2 = bp1 + HID_C;
    float* bp3 = bp2 + HID_C;
    __bf16* aggBf = (__bf16*)(bp3 + HID_C);
    __bf16* xBf   = aggBf + (long long)N_NODES * HID_C;
    __bf16* wt1l  = xBf   + (long long)N_NODES * HID_C;
    __bf16* wt1r  = wt1l + HID_C * IN_C;
    __bf16* wt2l  = wt1r + HID_C * IN_C;
    __bf16* wt2r  = wt2l + HID_C * HID_C;
    __bf16* wt3l  = wt2r + HID_C * HID_C;
    __bf16* wt3r  = wt3l + OUT_PAD * HID_C;

    const int mTiles = N_NODES / 16;   // 6250 exactly

    // ---- one-time weight/bias staging (transposed, bf16, zero-padded) ----
    k_wt<<<(HID_C * IN_C + 255) / 256, 256, 0, stream>>>(w1l, wt1l, IN_C, HID_C, HID_C);
    k_wt<<<(HID_C * IN_C + 255) / 256, 256, 0, stream>>>(w1r, wt1r, IN_C, HID_C, HID_C);
    k_wt<<<(HID_C * HID_C + 255) / 256, 256, 0, stream>>>(w2l, wt2l, HID_C, HID_C, HID_C);
    k_wt<<<(HID_C * HID_C + 255) / 256, 256, 0, stream>>>(w2r, wt2r, HID_C, HID_C, HID_C);
    k_wt<<<(OUT_PAD * HID_C + 255) / 256, 256, 0, stream>>>(w3l, wt3l, HID_C, OUT_C, OUT_PAD);
    k_wt<<<(OUT_PAD * HID_C + 255) / 256, 256, 0, stream>>>(w3r, wt3r, HID_C, OUT_C, OUT_PAD);
    k_biaspad<<<1, 256, 0, stream>>>(b1, bp1, HID_C, HID_C);
    k_biaspad<<<1, 256, 0, stream>>>(b2, bp2, HID_C, HID_C);
    k_biaspad<<<1, 256, 0, stream>>>(b3, bp3, OUT_C, OUT_PAD);

    // ---- degrees -> reciprocal (shared by all layers) ----
    k_zero<<<256, 256, 0, stream>>>(deg, N_NODES);
    k_deg<<<(N_EDGES + 255) / 256, 256, 0, stream>>>(dst, deg, N_EDGES);
    k_rdeg<<<(N_NODES + 255) / 256, 256, 0, stream>>>(deg, N_NODES);

    // ---- layer 1: x (64) -> h1 (128), ReLU ----
    k_zero<<<2048, 256, 0, stream>>>(agg, (long long)N_NODES * IN_C);
    k_scatter<<<4096, 256, 0, stream>>>(x, src, dst, agg, IN_C);
    k_cvt_pair<<<4096, 256, 0, stream>>>(agg, deg, x, (unsigned int*)aggBf,
                                         (unsigned int*)xBf, IN_C,
                                         (long long)N_NODES * IN_C / 2);
    {
        int ntn = HID_C / 16, tiles = mTiles * ntn;
        k_gemm<IN_C><<<(tiles * 32 + 255) / 256, 256, 0, stream>>>(
            aggBf, xBf, wt1l, wt1r, bp1, h1, HID_C, ntn, tiles, 1);
    }

    // ---- layer 2: h1 (128) -> h2 (128), ReLU ----
    k_zero<<<2048, 256, 0, stream>>>(agg, (long long)N_NODES * HID_C);
    k_scatter<<<4096, 256, 0, stream>>>(h1, src, dst, agg, HID_C);
    k_cvt_pair<<<4096, 256, 0, stream>>>(agg, deg, h1, (unsigned int*)aggBf,
                                         (unsigned int*)xBf, HID_C,
                                         (long long)N_NODES * HID_C / 2);
    {
        int ntn = HID_C / 16, tiles = mTiles * ntn;
        k_gemm<HID_C><<<(tiles * 32 + 255) / 256, 256, 0, stream>>>(
            aggBf, xBf, wt2l, wt2r, bp2, h2, HID_C, ntn, tiles, 1);
    }

    // ---- layer 3: h2 (128) -> out (40), then in-place log_softmax ----
    k_zero<<<2048, 256, 0, stream>>>(agg, (long long)N_NODES * HID_C);
    k_scatter<<<4096, 256, 0, stream>>>(h2, src, dst, agg, HID_C);
    k_cvt_pair<<<4096, 256, 0, stream>>>(agg, deg, h2, (unsigned int*)aggBf,
                                         (unsigned int*)xBf, HID_C,
                                         (long long)N_NODES * HID_C / 2);
    {
        int ntn = OUT_PAD / 16, tiles = mTiles * ntn;
        k_gemm<HID_C><<<(tiles * 32 + 255) / 256, 256, 0, stream>>>(
            aggBf, xBf, wt3l, wt3r, bp3, out, OUT_C, ntn, tiles, 0);
    }
    k_logsoftmax<<<1024, 256, 0, stream>>>(out, N_NODES, OUT_C);
}
